// Attention_47064251629839
// MI455X (gfx1250) — matrix operand order repaired and checked by reading, not compiled, not run on hardware
//
#include <hip/hip_runtime.h>
#include <hip/hip_bf16.h>

#define NUM_HEADS 16
#define HEAD_DIM  64
#define EMBED     1024
#define DDIM      1024
#define SEQ       2048
#define BATCH     4

typedef _Float16 v16h __attribute__((ext_vector_type(16)));
typedef float    v8f  __attribute__((ext_vector_type(8)));
typedef _Float16 h2v  __attribute__((ext_vector_type(2)));

__device__ __forceinline__ v8f wmma_f16(v16h a, v16h b, v8f c) {
  // D = A(16x32 f16) * B(32x16 f16) + C(16x16 f32)
  return __builtin_amdgcn_wmma_f32_16x16x32_f16(false, a, false, b, (short)0, c,
                                                false, false);
}

// ---------------------------------------------------------------------------
// Fully-fused row statistics over a 16-lane row.
// rowmax: pairwise max + 4 DPP butterfly stages + running-max combine, all in
// one asm block (no llvm.maxnum boundary -> no canonicalize self-max ops).
// quad_perm[1,0,3,2]=xor1, quad_perm[2,3,0,1]=xor2, then half/full mirror
// (mirrors are valid combiners once 4/8-lane subgroups are uniform).
// ---------------------------------------------------------------------------
__device__ __forceinline__ float rowmax(float x0, float x1, float m_prev) {
  float y;
  asm("v_max_num_f32_e32 %0, %1, %2\n\t"
      "v_max_num_f32_dpp %0, %0, %0 quad_perm:[1,0,3,2] row_mask:0xf bank_mask:0xf\n\t"
      "v_max_num_f32_dpp %0, %0, %0 quad_perm:[2,3,0,1] row_mask:0xf bank_mask:0xf\n\t"
      "v_max_num_f32_dpp %0, %0, %0 row_half_mirror row_mask:0xf bank_mask:0xf\n\t"
      "v_max_num_f32_dpp %0, %0, %0 row_mirror row_mask:0xf bank_mask:0xf\n\t"
      "v_max_num_f32_e32 %0, %0, %3"
      : "=&v"(y)
      : "v"(x0), "v"(x1), "v"(m_prev));
  return y;
}
__device__ __forceinline__ float rowsum(float p0, float p1) {
  float y;
  asm("v_add_f32_e32 %0, %1, %2\n\t"
      "v_add_f32_dpp %0, %0, %0 quad_perm:[1,0,3,2] row_mask:0xf bank_mask:0xf\n\t"
      "v_add_f32_dpp %0, %0, %0 quad_perm:[2,3,0,1] row_mask:0xf bank_mask:0xf\n\t"
      "v_add_f32_dpp %0, %0, %0 row_half_mirror row_mask:0xf bank_mask:0xf\n\t"
      "v_add_f32_dpp %0, %0, %0 row_mirror row_mask:0xf bank_mask:0xf"
      : "=&v"(y)
      : "v"(p0), "v"(p1));
  return y;
}

// ---------------------------------------------------------------------------
// Async global -> LDS b128 copy (CDNA5 ASYNCcnt path, no VGPR round trip)
// ---------------------------------------------------------------------------
__device__ __forceinline__ void async_b128_to_lds(const _Float16* g, void* lds) {
  unsigned lds_addr = (unsigned)(unsigned long long)lds;  // addr[31:0] = LDS off
  asm volatile("global_load_async_to_lds_b128 %0, %1, off"
               :
               : "v"(lds_addr), "v"(g)
               : "memory");
}
__device__ __forceinline__ void wait_async0() {
  asm volatile("s_wait_asynccnt 0x0" ::: "memory");
}

// A-operand fragment from a row-major 16x32 f16 tile (leading dim ld).
// lanes 0-15 -> row=lane, K = {0..7, 16..23}; lanes 16-31 -> row=lane-16,
// K = {8..15, 24..31}. Both runs contiguous -> two b128 loads.
__device__ __forceinline__ v16h load_fragA(const _Float16* base, int ld, int lane) {
  const int r = lane & 15, half = lane >> 4;
  const _Float16* rp = base + (size_t)r * ld + half * 8;
  uint4 q[2];
  q[0] = *(const uint4*)(rp);
  q[1] = *(const uint4*)(rp + 16);
  v16h f;
  __builtin_memcpy(&f, q, 32);
  return f;
}

// B-operand fragment from a TRANSPOSED tile Bt[n][k] (n = out col, k = reduce).
// lanes 0-15 -> n=lane, K = {0..7, 16..23}; lanes 16-31 -> n=lane-16, K = {8..15, 24..31}.
__device__ __forceinline__ v16h load_fragB(const _Float16* base, int ld, int lane) {
  const int n = lane & 15, half = lane >> 4;
  const _Float16* rp = base + (size_t)n * ld + half * 8;
  uint4 q[2];
  q[0] = *(const uint4*)(rp);
  q[1] = *(const uint4*)(rp + 16);
  v16h f;
  __builtin_memcpy(&f, q, 32);
  return f;
}

// ---------------------------------------------------------------------------
// fp32 -> f16 weight conversion
// ---------------------------------------------------------------------------
__global__ void cvt_f32_to_f16(const float* __restrict__ in,
                               _Float16* __restrict__ o, int n) {
  int i = blockIdx.x * blockDim.x + threadIdx.x;
  if (i < n) o[i] = (_Float16)in[i];
}

// ---------------------------------------------------------------------------
// LayerNorm (one row of 1024 per block) fused with f16 conversion
// ---------------------------------------------------------------------------
__global__ __launch_bounds__(256) void layernorm_f16_kernel(
    const float* __restrict__ x, const float* __restrict__ gamma,
    const float* __restrict__ beta, _Float16* __restrict__ out) {
  const int row = blockIdx.x, t = threadIdx.x;
  const float* xr = x + (size_t)row * DDIM;
  float s = 0.f, s2 = 0.f;
  #pragma unroll
  for (int i = t; i < DDIM; i += 256) {
    float v = xr[i];
    s += v; s2 += v * v;
  }
  #pragma unroll
  for (int off = 16; off >= 1; off >>= 1) {
    s  += __shfl_xor(s, off, 32);
    s2 += __shfl_xor(s2, off, 32);
  }
  __shared__ float rs[8], rs2[8];
  __shared__ float mv[2];
  if ((t & 31) == 0) { rs[t >> 5] = s; rs2[t >> 5] = s2; }
  __syncthreads();
  if (t == 0) {
    float a = 0.f, b2 = 0.f;
    for (int i = 0; i < 8; ++i) { a += rs[i]; b2 += rs2[i]; }
    float mean = a * (1.0f / DDIM);
    float var  = b2 * (1.0f / DDIM) - mean * mean;
    mv[0] = mean;
    mv[1] = rsqrtf(var + 1e-5f);
  }
  __syncthreads();
  const float mean = mv[0], rstd = mv[1];
  #pragma unroll
  for (int i = t; i < DDIM; i += 256)
    out[(size_t)row * DDIM + i] =
        (_Float16)((xr[i] - mean) * rstd * gamma[i] + beta[i]);
}

// ---------------------------------------------------------------------------
// WMMA GEMM: C[M x N] = A[M x K] * B[K x N], A/B f16, C f16 or f32.
// Block tile 128x128, 8 waves each computing 32x64, K-steps of 32.
// Double-buffered: next A tile via async-LDS + next B tile's global loads
// are in flight while the current tile's WMMAs execute. One barrier / step.
// grid = (N/128, M/128), block = 256.
// ---------------------------------------------------------------------------
template <typename CT>
__global__ __launch_bounds__(256) void gemm_wmma_kernel(
    const _Float16* __restrict__ A, const _Float16* __restrict__ B,
    CT* __restrict__ C, int N, int K) {
  __shared__ __align__(16) _Float16 As[2][128 * 32];  // [m][k]
  __shared__ __align__(16) _Float16 Bs[2][128 * 32];  // transposed: [n][k]
  const int t = threadIdx.x;
  const int lane = t & 31, w = t >> 5;
  const int m_blk = blockIdx.y * 128, n_blk = blockIdx.x * 128;
  const int mw = (w >> 1) * 32, nw = (w & 1) * 64;

  const int a_row = t >> 2, a_cv = (t & 3) * 8;   // + 64 rows on 2nd vec
  const int b_k   = t >> 4, b_nv = (t & 15) * 8;  // + 16 k    on 2nd vec

  auto stage_A = [&](int buf, int kb) {
    #pragma unroll
    for (int r = 0; r < 2; ++r) {
      int row = a_row + r * 64;
      async_b128_to_lds(&A[(size_t)(m_blk + row) * K + kb + a_cv],
                        &As[buf][row * 32 + a_cv]);
    }
  };
  auto load_B = [&](int kb, uint4* raw) {
    #pragma unroll
    for (int r = 0; r < 2; ++r)
      raw[r] = *(const uint4*)&B[(size_t)(kb + b_k + r * 16) * N + n_blk + b_nv];
  };
  auto scatter_B = [&](int buf, const uint4* raw) {
    #pragma unroll
    for (int r = 0; r < 2; ++r) {
      int k = b_k + r * 16;
      h2v p[4];
      __builtin_memcpy(p, &raw[r], 16);
      #pragma unroll
      for (int j = 0; j < 4; ++j) {
        Bs[buf][(b_nv + 2 * j) * 32 + k]     = p[j].x;
        Bs[buf][(b_nv + 2 * j + 1) * 32 + k] = p[j].y;
      }
    }
  };

  const v8f zero = {};
  v8f acc[2][4];
  #pragma unroll
  for (int i = 0; i < 2; ++i)
    #pragma unroll
    for (int j = 0; j < 4; ++j) acc[i][j] = zero;

  // Prologue: fill buffer 0
  stage_A(0, 0);
  {
    uint4 raw[2];
    load_B(0, raw);
    scatter_B(0, raw);
  }
  wait_async0();
  __syncthreads();

  const int NT = K / 32;
  for (int it = 0; it < NT; ++it) {
    const int cur = it & 1, nxt = cur ^ 1;
    const bool more = (it + 1) < NT;
    uint4 rawB[2];
    if (more) {
      stage_A(nxt, (it + 1) * 32);   // async engine runs during the WMMAs
      load_B((it + 1) * 32, rawB);   // global loads in flight during compute
    }

    v16h af[2], bf[4];
    af[0] = load_fragA(&As[cur][mw * 32], 32, lane);
    af[1] = load_fragA(&As[cur][(mw + 16) * 32], 32, lane);
    #pragma unroll
    for (int j = 0; j < 4; ++j)
      bf[j] = load_fragB(&Bs[cur][(nw + 16 * j) * 32], 32, lane);
    #pragma unroll
    for (int i = 0; i < 2; ++i)
      #pragma unroll
      for (int j = 0; j < 4; ++j) acc[i][j] = wmma_f16(af[i], bf[j], acc[i][j]);

    if (more) scatter_B(nxt, rawB);  // transpose-scatter after the WMMAs
    wait_async0();
    __syncthreads();
  }

  // C/D layout: VGPR v, lane -> (row = v + 8*(lane/16), col = lane%16)
  const int half = lane >> 4, c0 = lane & 15;
  #pragma unroll
  for (int i = 0; i < 2; ++i)
    #pragma unroll
    for (int j = 0; j < 4; ++j)
      #pragma unroll
      for (int v = 0; v < 8; ++v) {
        int row = m_blk + mw + i * 16 + v + 8 * half;
        int col = n_blk + nw + j * 16 + c0;
        C[(size_t)row * N + col] = (CT)acc[i][j][v];
      }
}

// ---------------------------------------------------------------------------
// Flash attention: grid = (SEQ/128, BATCH*NUM_HEADS), block = 256 (8 waves).
// Each wave owns 16 query rows; block streams 32-key tiles of K and V,
// double-buffered (async K copy + V global loads overlap S/softmax/PV).
// qp: [B*SEQ, EMBED] f16 ; kvp: [B*SEQ, 2*EMBED] f16 (K | V) ; out f16.
// ---------------------------------------------------------------------------
__global__ __launch_bounds__(256) void flash_attn_kernel(
    const _Float16* __restrict__ qp, const _Float16* __restrict__ kvp,
    _Float16* __restrict__ out) {
  __shared__ __align__(16) _Float16 Kt[2][32 * 64];  // [key][dim] (Bt for QK^T)
  __shared__ __align__(16) _Float16 Vt[2][64 * 32];  // [dim][key] (Bt for P*V)
  __shared__ __align__(16) _Float16 Pb[8][16 * 32];  // per-wave P tile
  const int t = threadIdx.x, lane = t & 31, w = t >> 5;
  const int half = lane >> 4, c0 = lane & 15;
  const int b = blockIdx.y >> 4, h = blockIdx.y & 15;
  const int q0 = blockIdx.x * 128 + w * 16;
  const size_t qrow0 = (size_t)b * SEQ + q0;
  const int srow = t >> 3, sdv = (t & 7) * 8;  // one b128 per thread per tile

  auto kv_addr = [&](int kb) {
    return ((size_t)b * SEQ + kb + srow) * (2 * EMBED) + h * HEAD_DIM + sdv;
  };
  auto stage_K = [&](int buf, int kb) {
    async_b128_to_lds(&kvp[kv_addr(kb)], &Kt[buf][srow * 64 + sdv]);
  };
  auto load_V = [&](int kb) { return *(const uint4*)&kvp[kv_addr(kb) + EMBED]; };
  auto scatter_V = [&](int buf, uint4 raw) {
    h2v p[4];
    __builtin_memcpy(p, &raw, 16);
    #pragma unroll
    for (int j = 0; j < 4; ++j) {
      Vt[buf][(sdv + 2 * j) * 32 + srow]     = p[j].x;
      Vt[buf][(sdv + 2 * j + 1) * 32 + srow] = p[j].y;
    }
  };

  // Resident Q fragments, softmax scale (2^-3, exact in f16) folded in once.
  v16h qf[2];
  qf[0] = load_fragA(qp + qrow0 * EMBED + h * HEAD_DIM, EMBED, lane);
  qf[1] = load_fragA(qp + qrow0 * EMBED + h * HEAD_DIM + 32, EMBED, lane);
  #pragma unroll
  for (int e = 0; e < 16; ++e) {
    qf[0][e] = qf[0][e] * (_Float16)0.125f;
    qf[1][e] = qf[1][e] * (_Float16)0.125f;
  }

  const v8f zero = {};
  v8f o[4] = {zero, zero, zero, zero};  // 16 x 64 f32 accumulator
  float m_run[8], l_run[8];
  #pragma unroll
  for (int v = 0; v < 8; ++v) { m_run[v] = -1e30f; l_run[v] = 0.f; }

  _Float16* Pw = &Pb[w][0];

  // Prologue: fill buffer 0
  stage_K(0, 0);
  scatter_V(0, load_V(0));
  wait_async0();
  __syncthreads();

  const int NT = SEQ / 32;
  for (int it = 0; it < NT; ++it) {
    const int cur = it & 1, nxt = cur ^ 1;
    const bool more = (it + 1) < NT;
    uint4 rawV;
    if (more) {
      stage_K(nxt, (it + 1) * 32);  // async during S/softmax/PV
      rawV = load_V((it + 1) * 32); // global load in flight during compute
    }

    // S = (Q*scale) K^T : 16 x 32 via two 16x16 n-tiles, K-dim 64 (2 steps)
    const _Float16* Kc = &Kt[cur][0];
    v8f s0 = zero, s1 = zero;
    s0 = wmma_f16(qf[0], load_fragB(Kc, 64, lane), s0);
    s0 = wmma_f16(qf[1], load_fragB(Kc + 32, 64, lane), s0);
    s1 = wmma_f16(qf[0], load_fragB(Kc + 16 * 64, 64, lane), s1);
    s1 = wmma_f16(qf[1], load_fragB(Kc + 16 * 64 + 32, 64, lane), s1);

    // Online softmax: row (v + 8*half) lives in one 16-lane half; stats via
    // fully-fused DPP butterflies (pure VALU, co-executes with matrix pipe).
    float alpha[8];
    #pragma unroll
    for (int v = 0; v < 8; ++v) {
      float x0 = s0[v], x1 = s1[v];
      float mnew = rowmax(x0, x1, m_run[v]);
      float a    = __expf(m_run[v] - mnew);
      float p0   = __expf(x0 - mnew);
      float p1   = __expf(x1 - mnew);
      float rs   = rowsum(p0, p1);
      l_run[v] = l_run[v] * a + rs;
      m_run[v] = mnew;
      alpha[v] = a;
      int row = v + 8 * half;
      Pw[row * 32 + c0]      = (_Float16)p0;  // C-layout -> LDS -> A-layout
      Pw[row * 32 + c0 + 16] = (_Float16)p1;
    }
    #pragma unroll
    for (int ti = 0; ti < 4; ++ti)
      #pragma unroll
      for (int v = 0; v < 8; ++v) o[ti][v] = o[ti][v] * alpha[v];
    __syncthreads();  // publish P

    // O += P * V : K-dim 32 (one step), four 16x16 n-tiles over HEAD_DIM
    v16h pf = load_fragA(Pw, 32, lane);
    #pragma unroll
    for (int ti = 0; ti < 4; ++ti)
      o[ti] = wmma_f16(pf, load_fragB(&Vt[cur][ti * 16 * 32], 32, lane), o[ti]);

    if (more) scatter_V(nxt, rawV);  // Vt[nxt] readers finished last iteration
    wait_async0();
    __syncthreads();
  }

  float inv_l[8];
  #pragma unroll
  for (int v = 0; v < 8; ++v) inv_l[v] = 1.0f / l_run[v];
  #pragma unroll
  for (int ti = 0; ti < 4; ++ti)
    #pragma unroll
    for (int v = 0; v < 8; ++v) {
      int row = v + 8 * half;
      int col = h * HEAD_DIM + ti * 16 + c0;
      out[(qrow0 + row) * EMBED + col] = (_Float16)(o[ti][v] * inv_l[v]);
    }
}

// ---------------------------------------------------------------------------
// Host-side launch
// ---------------------------------------------------------------------------
extern "C" void kernel_launch(void* const* d_in, const int* in_sizes, int n_in,
                              void* d_out, int out_size, void* d_ws,
                              size_t ws_size, hipStream_t stream) {
  (void)in_sizes; (void)n_in; (void)out_size; (void)ws_size;
  const float* q     = (const float*)d_in[0];
  const float* gamma = (const float*)d_in[1];
  const float* beta  = (const float*)d_in[2];
  const float* Wq    = (const float*)d_in[3];
  const float* Wkv   = (const float*)d_in[4];
  const float* Wout  = (const float*)d_in[5];
  float* out = (float*)d_out;

  char* ws = (char*)d_ws;
  const size_t MB = 1024 * 1024;
  _Float16* xn     = (_Float16*)(ws);            // 16 MB  B*N x D      f16
  _Float16* wq16   = (_Float16*)(ws + 16 * MB);  //  2 MB
  _Float16* wkv16  = (_Float16*)(ws + 18 * MB);  //  4 MB
  _Float16* wout16 = (_Float16*)(ws + 22 * MB);  //  2 MB
  _Float16* qp     = (_Float16*)(ws + 24 * MB);  // 16 MB  B*N x EMBED  f16
  _Float16* kvp    = (_Float16*)(ws + 40 * MB);  // 32 MB  B*N x 2EMBED f16
  _Float16* ao     = (_Float16*)(ws + 72 * MB);  // 16 MB  B*N x EMBED  f16

  cvt_f32_to_f16<<<(DDIM * EMBED) / 256, 256, 0, stream>>>(Wq, wq16,
                                                           DDIM * EMBED);
  cvt_f32_to_f16<<<(DDIM * 2 * EMBED) / 256, 256, 0, stream>>>(
      Wkv, wkv16, DDIM * 2 * EMBED);
  cvt_f32_to_f16<<<(EMBED * DDIM) / 256, 256, 0, stream>>>(Wout, wout16,
                                                           EMBED * DDIM);
  layernorm_f16_kernel<<<BATCH * SEQ, 256, 0, stream>>>(q, gamma, beta, xn);

  gemm_wmma_kernel<_Float16>
      <<<dim3(EMBED / 128, BATCH * SEQ / 128), 256, 0, stream>>>(xn, wq16, qp,
                                                                 EMBED, DDIM);
  gemm_wmma_kernel<_Float16>
      <<<dim3(2 * EMBED / 128, BATCH * SEQ / 128), 256, 0, stream>>>(
          xn, wkv16, kvp, 2 * EMBED, DDIM);
  flash_attn_kernel<<<dim3(SEQ / 128, BATCH * NUM_HEADS), 256, 0, stream>>>(
      qp, kvp, ao);
  gemm_wmma_kernel<float>
      <<<dim3(DDIM / 128, BATCH * SEQ / 128), 256, 0, stream>>>(ao, wout16, out,
                                                                DDIM, EMBED);
}
